// SO3HamNODE_10763188043881
// MI455X (gfx1250) — matrix operand order repaired and checked by reading, not compiled, not run on hardware
//
#include <hip/hip_runtime.h>
#include <hip/hip_bf16.h>
#include <math.h>

typedef __attribute__((ext_vector_type(16))) _Float16 v16h;
typedef __attribute__((ext_vector_type(8)))  _Float16 v8h;
typedef __attribute__((ext_vector_type(8)))  float    v8f;

#define HID   300
#define HP    320          // padded hidden (M-net / G-net)
#define NTH   (HP/16)      // 20 N-tiles
#define KCH   (HP/32)      // 10 K-chunks
#define VHID  50
#define VP    64           // padded V-net hidden
#define NTV   (VP/16)      // 4
#define KCV   (VP/32)      // 2
#define EPSV  0.01f
#define ROWS  16
#define TPB   128
#define NACC  5            // N-tiles (accumulators) per wave on 320-wide layers

// ---------------------------------------------------------------------------
// Weight packing: f32 [R x C] (optionally transposed) -> f16 padded [Kp x Np],
// pre-swizzled for WMMA B in this file's contiguous k order (same as load_a):
//   lane = (n%16) + 16*((k%32)>=16),  element = k%16,
//   dst[((kc*NT + nt)*32 + lane)*16 + e]
// ---------------------------------------------------------------------------
__global__ void pack_weight_k(const float* __restrict__ src, _Float16* __restrict__ dst,
                              int R, int C, int Kp, int Np, int trans) {
  const int NT = Np >> 4;
  const int total = Kp * Np;
  for (int idx = blockIdx.x * blockDim.x + threadIdx.x; idx < total;
       idx += gridDim.x * blockDim.x) {
    const int k = idx / Np;
    const int n = idx - k * Np;
    float v = 0.f;
    if (!trans) { if (k < R && n < C) v = src[(size_t)k * C + n]; }
    else        { if (k < C && n < R) v = src[(size_t)n * C + k]; }
    const int kc = k >> 5, kr = k & 31, nt = n >> 4, nn = n & 15;
    const int lane = nn + ((kr & 16) ? 16 : 0);
    const int e = kr & 15;
    dst[((size_t)(kc * NT + nt) * 32 + lane) * 16 + e] = (_Float16)v;
  }
}

// zero-padded f32 bias copy (removes per-tile range guards in the main kernel)
__global__ void pack_bias_k(const float* __restrict__ src, float* __restrict__ dst,
                            int n, int Np) {
  for (int i = blockIdx.x * blockDim.x + threadIdx.x; i < Np; i += gridDim.x * blockDim.x)
    dst[i] = (i < n) ? src[i] : 0.f;
}

// ---------------------------------------------------------------------------
// WMMA helpers
// ---------------------------------------------------------------------------

// A-fragment: two contiguous 16B-aligned 16-byte runs from LDS (-> ds_load_b128 x2)
static __device__ __forceinline__ v16h load_a(const _Float16* __restrict__ act, int stride,
                                              int kc, int lane) {
  const int m = lane & 15;
  const int g = (lane >> 4) << 4;   // 0 (lanes 0-15) or 16 (lanes 16-31)
  const _Float16* ap = act + m * stride + kc * 32;
  v8h lo = *(const v8h*)(ap + g);
  v8h hi = *(const v8h*)(ap + g + 8);
  v16h a;
#pragma unroll
  for (int i = 0; i < 8; ++i) { a[i] = lo[i]; a[8 + i] = hi[i]; }
  return a;
}

// Strip GEMM: one wave computes NA adjacent 16x16 output tiles (nt0..nt0+NA-1).
// A loaded once per k-chunk; NA independent accumulators keep the XDL pipe fed.
template <int NA>
static __device__ __forceinline__ void strip_gemm(const _Float16* __restrict__ act, int stride,
                                                  const _Float16* __restrict__ w, int NT, int nt0,
                                                  int kchunks, int lane, v8f* acc) {
  const v8f z = {};
#pragma unroll
  for (int t = 0; t < NA; ++t) acc[t] = z;
  for (int kc = 0; kc < kchunks; ++kc) {
    const v16h a = load_a(act, stride, kc, lane);
    v16h b[NA];
#pragma unroll
    for (int t = 0; t < NA; ++t)
      b[t] = *(const v16h*)(w + ((size_t)(kc * NT + nt0 + t) * 32 + lane) * 16);
#pragma unroll
    for (int t = 0; t < NA; ++t)
      acc[t] = __builtin_amdgcn_wmma_f32_16x16x32_f16(false, a, false, b[t], (short)0,
                                                      acc[t], false, false);
  }
}

// Single-tile GEMM (N=16 output layers, V-net tiles)
static __device__ __forceinline__ v8f tile_gemm(const _Float16* __restrict__ act, int stride,
                                                const _Float16* __restrict__ w, int NT, int nt,
                                                int kchunks, int lane) {
  v8f c = {};
  for (int kc = 0; kc < kchunks; ++kc) {
    const v16h a = load_a(act, stride, kc, lane);
    const v16h b = *(const v16h*)(w + ((size_t)(kc * NT + nt) * 32 + lane) * 16);
    c = __builtin_amdgcn_wmma_f32_16x16x32_f16(false, a, false, b, (short)0, c, false, false);
  }
  return c;
}

// branchless tanh: hardware exp + raw v_rcp_f32 (result truncated to f16 anyway).
// Saturates correctly for |x| large (exp->inf => 1, exp->0 => -1).
static __device__ __forceinline__ float fast_tanh(float x) {
  const float e = __expf(2.f * x);
  return 1.f - 2.f * __builtin_amdgcn_rcpf(e + 1.f);
}

// store tanh(c + bias) as f16 into LDS activation buffer (bias is zero-padded)
static __device__ __forceinline__ void store_tanh16(v8f c, _Float16* dst, int stride, int n0,
                                                    const float* __restrict__ bias, int lane) {
  const int n = n0 + (lane & 15);
  const float bv = bias[n];
  const int mb = (lane < 16) ? 0 : 8;
#pragma unroll
  for (int j = 0; j < 8; ++j) dst[(mb + j) * stride + n] = (_Float16)fast_tanh(c[j] + bv);
}

// store (1 - h^2) * c as f16 (tanh' chain for VJP / JVP)
static __device__ __forceinline__ void store_dtanh16(v8f c, const _Float16* __restrict__ hbuf,
                                                     _Float16* dst, int stride, int n0, int lane) {
  const int n = n0 + (lane & 15);
  const int mb = (lane < 16) ? 0 : 8;
#pragma unroll
  for (int j = 0; j < 8; ++j) {
    const float h = (float)hbuf[(mb + j) * stride + n];
    dst[(mb + j) * stride + n] = (_Float16)((1.f - h * h) * c[j]);
  }
}

// raw f32 store of one 16x16 tile
static __device__ __forceinline__ void store_f32(v8f c, float* dst, int lane) {
  const int n = lane & 15;
  const int mb = (lane < 16) ? 0 : 8;
#pragma unroll
  for (int j = 0; j < 8; ++j) dst[(mb + j) * 16 + n] = c[j];
}

static __device__ __forceinline__ void cross3(const float* a, const float* b, float* c) {
  c[0] = a[1] * b[2] - a[2] * b[1];
  c[1] = a[2] * b[0] - a[0] * b[2];
  c[2] = a[0] * b[1] - a[1] * b[0];
}

struct FusedArgs {
  const float* x;
  const _Float16 *MW1, *MW2, *MW3, *MW4, *MW4T, *MW3T, *MW2T, *MW1T;
  const _Float16 *VW1, *VW2, *VW2T, *VW1T;
  const _Float16 *GW1, *GW2, *GW3;
  const float *Mb1, *Mb2, *Mb3, *Vb1, *Vb2, *VW3v, *Gb1, *Gb2;  // zero-padded f32
  float* out;
  int nrows;
};

// layer driver macros: 320-wide layers, NACC tiles per wave, contiguous strip
#define LAYER_TANH(IN, INSTR, W, KC, OUT, BIAS)                                  \
  {                                                                              \
    v8f acc[NACC];                                                               \
    strip_gemm<NACC>(IN, INSTR, W, NTH, wv * NACC, KC, lane, acc);               \
    _Pragma("unroll") for (int t = 0; t < NACC; ++t)                             \
        store_tanh16(acc[t], OUT, HP, (wv * NACC + t) * 16, BIAS, lane);         \
  }                                                                              \
  __syncthreads();

#define LAYER_DTANH(IN, INSTR, W, KC, HBUF, OUT)                                 \
  {                                                                              \
    v8f acc[NACC];                                                               \
    strip_gemm<NACC>(IN, INSTR, W, NTH, wv * NACC, KC, lane, acc);               \
    _Pragma("unroll") for (int t = 0; t < NACC; ++t)                             \
        store_dtanh16(acc[t], HBUF, OUT, HP, (wv * NACC + t) * 16, lane);        \
  }                                                                              \
  __syncthreads();

// ---------------------------------------------------------------------------
// Fused SO3-HamNODE forward: 16 batch rows per workgroup, 4 waves.
// ---------------------------------------------------------------------------
__global__ __launch_bounds__(TPB) void so3_fused_k(FusedArgs A) {
  __shared__ __align__(16) _Float16 sh1[ROWS * HP], sh2[ROWS * HP], sh3[ROWS * HP];
  __shared__ __align__(16) _Float16 sbA[ROWS * HP], sbB[ROWS * HP];   // ping-pong
  __shared__ __align__(16) _Float16 sqp[ROWS * 32];                   // q pad; later dq pad
  __shared__ __align__(16) _Float16 sdlo[ROWS * 32];                  // dlo pad (K=32)
  __shared__ __align__(16) _Float16 sv1[ROWS * VP], sv2[ROWS * VP];   // V-net
  __shared__ float fA[ROWS * 16], fB[ROWS * 16], fC[ROWS * 16];       // 16x16 f32 tiles
  __shared__ float r_q[ROWS * 9], r_qd[ROWS * 3], r_u[ROWS * 3];
  __shared__ float r_p[ROWS * 3], r_dHdp[ROWS * 3], r_dq[ROWS * 9], r_dp[ROWS * 3];
  __shared__ float r_L[ROWS * 9], r_Mi[ROWS * 9];

  const int tid = threadIdx.x, lane = tid & 31, wv = tid >> 5;
  const int row0 = blockIdx.x * ROWS;
  if (row0 >= A.nrows) return;

  // ---- stage 0: load x tile -> row scalars + qpad (f16, K padded to 32) ----
  for (int idx = tid; idx < ROWS * 15; idx += TPB) {
    const int r = idx / 15, c = idx - r * 15;
    const float v = A.x[(size_t)(row0 + r) * 15 + c];
    if (c < 9) r_q[r * 9 + c] = v;
    else if (c < 12) r_qd[r * 3 + (c - 9)] = v;
    else r_u[r * 3 + (c - 12)] = v;
  }
  for (int idx = tid; idx < ROWS * 32; idx += TPB) {
    const int r = idx >> 5, k = idx & 31;
    sqp[idx] = (k < 9) ? (_Float16)A.x[(size_t)(row0 + r) * 15 + k] : (_Float16)0.f;
  }
  __syncthreads();

  // ---- stage 1: M-net forward (keep h1,h2,h3 resident) ----
  LAYER_TANH(sqp, 32, A.MW1, 1,   sh1, A.Mb1)
  LAYER_TANH(sh1, HP, A.MW2, KCH, sh2, A.Mb2)
  LAYER_TANH(sh2, HP, A.MW3, KCH, sh3, A.Mb3)
  if (wv == 0) store_f32(tile_gemm(sh3, HP, A.MW4, 1, 0, KCH, lane), fA, lane);  // lo
  __syncthreads();

  // ---- stage 2: per-row 3x3: L, Minv=LL^T+eps, p=Minv^-1 qd, dHdp, dlo ----
  if (tid < ROWS) {
    const int r = tid;
    float L[9] = {0};
    L[0] = fA[r * 16 + 0]; L[4] = fA[r * 16 + 1]; L[8] = fA[r * 16 + 2];
    L[3] = fA[r * 16 + 3]; L[6] = fA[r * 16 + 4]; L[7] = fA[r * 16 + 5];
    float Mi[9];
#pragma unroll
    for (int i = 0; i < 3; ++i)
#pragma unroll
      for (int j = 0; j < 3; ++j) {
        float s = 0.f;
        for (int k = 0; k < 3; ++k) s += L[i * 3 + k] * L[j * 3 + k];
        Mi[i * 3 + j] = s + ((i == j) ? EPSV : 0.f);
      }
    const float c00 = Mi[4] * Mi[8] - Mi[5] * Mi[7];
    const float c01 = Mi[5] * Mi[6] - Mi[3] * Mi[8];
    const float c02 = Mi[3] * Mi[7] - Mi[4] * Mi[6];
    const float det = Mi[0] * c00 + Mi[1] * c01 + Mi[2] * c02;
    const float id = 1.f / det;
    float Inv[9];
    Inv[0] = c00 * id;                                Inv[1] = (Mi[2] * Mi[7] - Mi[1] * Mi[8]) * id;
    Inv[2] = (Mi[1] * Mi[5] - Mi[2] * Mi[4]) * id;    Inv[3] = c01 * id;
    Inv[4] = (Mi[0] * Mi[8] - Mi[2] * Mi[6]) * id;    Inv[5] = (Mi[2] * Mi[3] - Mi[0] * Mi[5]) * id;
    Inv[6] = c02 * id;                                Inv[7] = (Mi[1] * Mi[6] - Mi[0] * Mi[7]) * id;
    Inv[8] = (Mi[0] * Mi[4] - Mi[1] * Mi[3]) * id;
    float p[3], dHdp[3];
#pragma unroll
    for (int i = 0; i < 3; ++i)
      p[i] = Inv[i * 3 + 0] * r_qd[r * 3 + 0] + Inv[i * 3 + 1] * r_qd[r * 3 + 1] +
             Inv[i * 3 + 2] * r_qd[r * 3 + 2];
#pragma unroll
    for (int i = 0; i < 3; ++i)
      dHdp[i] = Mi[i * 3 + 0] * p[0] + Mi[i * 3 + 1] * p[1] + Mi[i * 3 + 2] * p[2];
    // VJP seed through LL^T: dL = (p p^T) L  -> 6 packed entries
    float s[3];
#pragma unroll
    for (int j = 0; j < 3; ++j) s[j] = p[0] * L[0 * 3 + j] + p[1] * L[1 * 3 + j] + p[2] * L[2 * 3 + j];
    const float dlo[6] = {p[0] * s[0], p[1] * s[1], p[2] * s[2], p[1] * s[0], p[2] * s[0], p[2] * s[1]};
#pragma unroll
    for (int k = 0; k < 32; ++k) sdlo[r * 32 + k] = (k < 6) ? (_Float16)dlo[k] : (_Float16)0.f;
#pragma unroll
    for (int i = 0; i < 9; ++i) { r_L[r * 9 + i] = L[i]; r_Mi[r * 9 + i] = Mi[i]; }
#pragma unroll
    for (int i = 0; i < 3; ++i) { r_p[r * 3 + i] = p[i]; r_dHdp[r * 3 + i] = dHdp[i]; }
  }
  __syncthreads();

  // ---- stage 3: V-net forward + backward (dV/dq) ----
  for (int nt = wv; nt < NTV; nt += 4)
    store_tanh16(tile_gemm(sqp, 32, A.VW1, NTV, nt, 1, lane), sv1, VP, nt * 16, A.Vb1, lane);
  __syncthreads();
  for (int nt = wv; nt < NTV; nt += 4)
    store_tanh16(tile_gemm(sv1, VP, A.VW2, NTV, nt, KCV, lane), sv2, VP, nt * 16, A.Vb2, lane);
  __syncthreads();
  for (int idx = tid; idx < ROWS * VP; idx += TPB) {   // delta2 = (1-h2^2) * VW3 (in place)
    const int n = idx & (VP - 1);
    const float h = (float)sv2[idx];
    sv2[idx] = (_Float16)((1.f - h * h) * A.VW3v[n]);  // VW3 zero-padded to 64
  }
  __syncthreads();
  for (int nt = wv; nt < NTV; nt += 4)                 // delta1 = (1-h1^2) . (delta2 @ VW2^T)
    store_dtanh16(tile_gemm(sv2, VP, A.VW2T, NTV, nt, KCV, lane), sv1, sv1, VP, nt * 16, lane);
  __syncthreads();
  if (wv == 0) store_f32(tile_gemm(sv1, VP, A.VW1T, 1, 0, KCV, lane), fB, lane); // dVdq
  __syncthreads();

  // ---- stage 4: M-net VJP: dHdq_M = backprop(dlo) ----
  LAYER_DTANH(sdlo, 32, A.MW4T, 1,   sh3, sbA)
  LAYER_DTANH(sbA,  HP, A.MW3T, KCH, sh2, sbB)
  LAYER_DTANH(sbB,  HP, A.MW2T, KCH, sh1, sbA)
  if (wv == 0) store_f32(tile_gemm(sbA, HP, A.MW1T, 1, 0, KCH, lane), fC, lane); // dHdq_M
  __syncthreads();

  // ---- stage 5: G-net forward (uses sqp = q, bufs free now) ----
  LAYER_TANH(sqp, 32, A.GW1, 1,   sbA, A.Gb1)
  LAYER_TANH(sbA, HP, A.GW2, KCH, sbB, A.Gb2)
  if (wv == 0) store_f32(tile_gemm(sbB, HP, A.GW3, 1, 0, KCH, lane), fA, lane);  // go (9 valid)
  __syncthreads();

  // ---- stage 6: per-row dq, F, dp; write dq-pad into sqp for the JVP ----
  if (tid < ROWS) {
    const int r = tid;
    float dHdq[9];
#pragma unroll
    for (int i = 0; i < 9; ++i) dHdq[i] = fC[r * 16 + i] + fB[r * 16 + i];
    const float* q = &r_q[r * 9];
    const float* dHdp = &r_dHdp[r * 3];
    float dq[9];
    cross3(q + 0, dHdp, dq + 0);
    cross3(q + 3, dHdp, dq + 3);
    cross3(q + 6, dHdp, dq + 6);
#pragma unroll
    for (int i = 0; i < 9; ++i) r_dq[r * 9 + i] = dq[i];
#pragma unroll
    for (int k = 0; k < 32; ++k) sqp[r * 32 + k] = (k < 9) ? (_Float16)dq[k] : (_Float16)0.f;
    float F[3];
#pragma unroll
    for (int i = 0; i < 3; ++i)
      F[i] = fA[r * 16 + 3 * i + 0] * r_u[r * 3 + 0] + fA[r * 16 + 3 * i + 1] * r_u[r * 3 + 1] +
             fA[r * 16 + 3 * i + 2] * r_u[r * 3 + 2];
    float acc[3], t[3];
    cross3(&r_p[r * 3], dHdp, acc);
#pragma unroll
    for (int i = 0; i < 3; ++i) acc[i] += F[i];
    cross3(q + 0, dHdq + 0, t); acc[0] += t[0]; acc[1] += t[1]; acc[2] += t[2];
    cross3(q + 3, dHdq + 3, t); acc[0] += t[0]; acc[1] += t[1]; acc[2] += t[2];
    cross3(q + 6, dHdq + 6, t); acc[0] += t[0]; acc[1] += t[1]; acc[2] += t[2];
#pragma unroll
    for (int i = 0; i < 3; ++i) r_dp[r * 3 + i] = acc[i];
  }
  __syncthreads();

  // ---- stage 7: M-net JVP along dq (tangents through stored h1,h2,h3) ----
  LAYER_DTANH(sqp, 32, A.MW1, 1,   sh1, sbA)
  LAYER_DTANH(sbA, HP, A.MW2, KCH, sh2, sbB)
  LAYER_DTANH(sbB, HP, A.MW3, KCH, sh3, sbA)
  if (wv == 0) store_f32(tile_gemm(sbA, HP, A.MW4, 1, 0, KCH, lane), fB, lane);  // dlo_t
  __syncthreads();

  // ---- stage 8: per-row dMinv/dt, ddq, final store ----
  if (tid < ROWS) {
    const int r = tid;
    float dLt[9] = {0};
    dLt[0] = fB[r * 16 + 0]; dLt[4] = fB[r * 16 + 1]; dLt[8] = fB[r * 16 + 2];
    dLt[3] = fB[r * 16 + 3]; dLt[6] = fB[r * 16 + 4]; dLt[7] = fB[r * 16 + 5];
    const float* L = &r_L[r * 9];
    const float* Mi = &r_Mi[r * 9];
    const float* p = &r_p[r * 3];
    const float* dp = &r_dp[r * 3];
    float dMt[9];
#pragma unroll
    for (int i = 0; i < 3; ++i)
#pragma unroll
      for (int j = 0; j < 3; ++j) {
        float sum = 0.f;
        for (int k = 0; k < 3; ++k)
          sum += dLt[i * 3 + k] * L[j * 3 + k] + L[i * 3 + k] * dLt[j * 3 + k];
        dMt[i * 3 + j] = sum;
      }
    float ddq[3];
#pragma unroll
    for (int i = 0; i < 3; ++i)
      ddq[i] = Mi[i * 3 + 0] * dp[0] + Mi[i * 3 + 1] * dp[1] + Mi[i * 3 + 2] * dp[2] +
               dMt[i * 3 + 0] * p[0] + dMt[i * 3 + 1] * p[1] + dMt[i * 3 + 2] * p[2];
    const size_t ob = (size_t)(row0 + r) * 15;
#pragma unroll
    for (int i = 0; i < 9; ++i) A.out[ob + i] = r_dq[r * 9 + i];
#pragma unroll
    for (int i = 0; i < 3; ++i) A.out[ob + 9 + i] = ddq[i];
#pragma unroll
    for (int i = 0; i < 3; ++i) A.out[ob + 12 + i] = 0.f;
  }
}

// ---------------------------------------------------------------------------
extern "C" void kernel_launch(void* const* d_in, const int* in_sizes, int n_in,
                              void* d_out, int out_size, void* d_ws, size_t ws_size,
                              hipStream_t stream) {
  const float* x   = (const float*)d_in[0];
  const float* MW1 = (const float*)d_in[1];  const float* Mb1 = (const float*)d_in[2];
  const float* MW2 = (const float*)d_in[3];  const float* Mb2 = (const float*)d_in[4];
  const float* MW3 = (const float*)d_in[5];  const float* Mb3 = (const float*)d_in[6];
  const float* MW4 = (const float*)d_in[7];
  const float* VW1 = (const float*)d_in[8];  const float* Vb1 = (const float*)d_in[9];
  const float* VW2 = (const float*)d_in[10]; const float* Vb2 = (const float*)d_in[11];
  const float* VW3 = (const float*)d_in[12];
  const float* GW1 = (const float*)d_in[13]; const float* Gb1 = (const float*)d_in[14];
  const float* GW2 = (const float*)d_in[15]; const float* Gb2 = (const float*)d_in[16];
  const float* GW3 = (const float*)d_in[17];

  char* ws = (char*)d_ws;
  size_t off = 0;
  auto alloc16 = [&](size_t elems) { _Float16* p = (_Float16*)(ws + off);
                                     off += ((elems * sizeof(_Float16) + 15) & ~(size_t)15);
                                     return p; };
  auto alloc32 = [&](size_t elems) { float* p = (float*)(ws + off);
                                     off += ((elems * sizeof(float) + 15) & ~(size_t)15);
                                     return p; };
  _Float16* pMW1  = alloc16(32 * 320);
  _Float16* pMW2  = alloc16(320 * 320);
  _Float16* pMW3  = alloc16(320 * 320);
  _Float16* pMW4  = alloc16(320 * 16);
  _Float16* pMW4T = alloc16(32 * 320);
  _Float16* pMW3T = alloc16(320 * 320);
  _Float16* pMW2T = alloc16(320 * 320);
  _Float16* pMW1T = alloc16(320 * 16);
  _Float16* pVW1  = alloc16(32 * 64);
  _Float16* pVW2  = alloc16(64 * 64);
  _Float16* pVW2T = alloc16(64 * 64);
  _Float16* pVW1T = alloc16(64 * 16);
  _Float16* pGW1  = alloc16(32 * 320);
  _Float16* pGW2  = alloc16(320 * 320);
  _Float16* pGW3  = alloc16(320 * 16);
  float* bMb1 = alloc32(HP); float* bMb2 = alloc32(HP); float* bMb3 = alloc32(HP);
  float* bGb1 = alloc32(HP); float* bGb2 = alloc32(HP);
  float* bVb1 = alloc32(VP); float* bVb2 = alloc32(VP); float* bVW3 = alloc32(VP);
  (void)ws_size; (void)n_in; (void)out_size;

#define PACK(dst, src, R, C, Kp, Np, tr) \
  pack_weight_k<<<(((Kp) * (Np)) + 255) / 256, 256, 0, stream>>>(src, dst, R, C, Kp, Np, tr)
  PACK(pMW1,  MW1, 9,   300, 32,  320, 0);
  PACK(pMW2,  MW2, 300, 300, 320, 320, 0);
  PACK(pMW3,  MW3, 300, 300, 320, 320, 0);
  PACK(pMW4,  MW4, 300, 6,   320, 16,  0);
  PACK(pMW4T, MW4, 300, 6,   32,  320, 1);
  PACK(pMW3T, MW3, 300, 300, 320, 320, 1);
  PACK(pMW2T, MW2, 300, 300, 320, 320, 1);
  PACK(pMW1T, MW1, 9,   300, 320, 16,  1);
  PACK(pVW1,  VW1, 9,   50,  32,  64,  0);
  PACK(pVW2,  VW2, 50,  50,  64,  64,  0);
  PACK(pVW2T, VW2, 50,  50,  64,  64,  1);
  PACK(pVW1T, VW1, 9,   50,  64,  16,  1);
  PACK(pGW1,  GW1, 9,   300, 32,  320, 0);
  PACK(pGW2,  GW2, 300, 300, 320, 320, 0);
  PACK(pGW3,  GW3, 300, 9,   320, 16,  0);
#undef PACK
#define PACKB(dst, src, n, Np) pack_bias_k<<<1, 256, 0, stream>>>(src, dst, n, Np)
  PACKB(bMb1, Mb1, HID, HP);  PACKB(bMb2, Mb2, HID, HP);  PACKB(bMb3, Mb3, HID, HP);
  PACKB(bGb1, Gb1, HID, HP);  PACKB(bGb2, Gb2, HID, HP);
  PACKB(bVb1, Vb1, VHID, VP); PACKB(bVb2, Vb2, VHID, VP); PACKB(bVW3, VW3, VHID, VP);
#undef PACKB

  FusedArgs fa;
  fa.x = x;
  fa.MW1 = pMW1; fa.MW2 = pMW2; fa.MW3 = pMW3; fa.MW4 = pMW4;
  fa.MW4T = pMW4T; fa.MW3T = pMW3T; fa.MW2T = pMW2T; fa.MW1T = pMW1T;
  fa.VW1 = pVW1; fa.VW2 = pVW2; fa.VW2T = pVW2T; fa.VW1T = pVW1T;
  fa.GW1 = pGW1; fa.GW2 = pGW2; fa.GW3 = pGW3;
  fa.Mb1 = bMb1; fa.Mb2 = bMb2; fa.Mb3 = bMb3;
  fa.Vb1 = bVb1; fa.Vb2 = bVb2; fa.VW3v = bVW3;
  fa.Gb1 = bGb1; fa.Gb2 = bGb2;
  fa.out = (float*)d_out;
  fa.nrows = in_sizes[0] / 15;

  const int grid = (fa.nrows + ROWS - 1) / ROWS;
  so3_fused_k<<<grid, TPB, 0, stream>>>(fa);
}
